// xLSTM_24885040513348
// MI455X (gfx1250) — matrix operand order repaired and checked by reading, not compiled, not run on hardware
//
#include <hip/hip_runtime.h>
#include <stddef.h>

// ---------------- problem constants ----------------
#define LL   2
#define DD   512
#define HH   8
#define DHH  64
#define PP   1024
#define HDD  512
#define KC   4
#define SS   128
#define BB   8
#define VV   32000
#define EPSF 1e-5f

typedef __attribute__((ext_vector_type(16))) __bf16 v16bf;
typedef __attribute__((ext_vector_type(8)))  float  v8f;

__device__ inline float sigmoidf_(float x) { return 1.0f / (1.0f + __expf(-x)); }

__device__ inline unsigned short bf16_bits(float f) {
    union { float f; unsigned u; } v; v.f = f;
    unsigned r = v.u + 0x7fffu + ((v.u >> 16) & 1u);   // round-to-nearest-even
    return (unsigned short)(r >> 16);
}

// ---------------- embedding gather (float4 copy) ----------------
__global__ __launch_bounds__(128) void embed_kernel(const int* __restrict__ tokens,
                                                    const float* __restrict__ emb,
                                                    float* __restrict__ xbuf) {
    int rb = blockIdx.x;                 // 0..S*B-1
    int tok = tokens[rb];
    const float4* src = (const float4*)(emb + (size_t)tok * DD);
    float4* dst = (float4*)(xbuf + (size_t)rb * DD);
    dst[threadIdx.x] = src[threadIdx.x];     // 128 threads x float4 == 512 floats
}

// ---------------- recurrent scan ----------------
struct ScanParams {
    const float *ln_g, *ln_b, *gn_g, *gn_b;
    const float *w_ul, *b_ul, *w_ur, *b_ur, *w_dn, *b_dn;
    const float *conv_w, *conv_b, *skip_w;
    const float *w_i, *b_i, *w_f, *b_f, *w_o, *b_o;
    const float *w_q, *b_q, *w_k, *b_k, *w_v, *b_v;
    float *xbuf, *outbuf, *cst, *nst, *mst;
    float *out_c, *out_n, *out_m;
};

__device__ inline float blockSum512(float v, float* red, int tid) {
#pragma unroll
    for (int o = 16; o > 0; o >>= 1) v += __shfl_down(v, o, 32);
    if ((tid & 31) == 0) red[tid >> 5] = v;
    __syncthreads();
    if (tid == 0) {
        float t = 0.f;
#pragma unroll
        for (int i = 0; i < 16; ++i) t += red[i];
        red[0] = t;
    }
    __syncthreads();
    float tot = red[0];
    __syncthreads();
    return tot;
}

__device__ inline float dot4(float4 a, float4 b, float s) {
    s = fmaf(a.x, b.x, s);
    s = fmaf(a.y, b.y, s);
    s = fmaf(a.z, b.z, s);
    s = fmaf(a.w, b.w, s);
    return s;
}

// wave-per-output GEMV: float4 weight reads (coalesced b128), wave32 shuffle reduce
__device__ inline void gemv16(const float* __restrict__ W, const float* __restrict__ bias,
                              const float* __restrict__ xin, float* __restrict__ xout,
                              int N, int Kd, int wave, int lane, float scale) {
    const float4* xi4 = (const float4*)xin;
    const int K4 = Kd >> 2;
    for (int j = wave; j < N; j += 16) {
        const float4* wr4 = (const float4*)(W + (size_t)j * Kd);
        float s = 0.f;
        for (int p = lane; p < K4; p += 32) s = dot4(wr4[p], xi4[p], s);
#pragma unroll
        for (int o = 16; o > 0; o >>= 1) s += __shfl_down(s, o, 32);
        if (lane == 0) xout[j] = (s + bias[j]) * scale;
    }
}

__global__ __launch_bounds__(512) void scan_kernel(ScanParams pr) {
    __shared__ __align__(16) float s_inp[DD], s_xn[PP];
    __shared__ __align__(16) float s_xt[PP], s_xc[PP];
    __shared__ __align__(16) float s_rt[HDD], s_q[HDD], s_k[HDD], s_v[HDD], s_ot[HDD];
    __shared__ __align__(16) float s_num[HDD], s_h[HDD];
    __shared__ float s_red[16];
    __shared__ float s_it[HH], s_ft[HH], s_ie[HH], s_fe[HH];
    __shared__ float s_den[HH], s_mu[HH], s_var[HH];

    const int b    = blockIdx.x;
    const int tid  = threadIdx.x;
    const int wave = tid >> 5;
    const int lane = tid & 31;

    // ----- init recurrent state for this batch element -----
    {
        const size_t CB = (size_t)HH * DHH * DHH;     // per (l,b)
        const size_t NB = (size_t)HH * DHH;
        for (int l = 0; l < LL; ++l) {
            float4* c4 = (float4*)(pr.cst + ((size_t)l * BB + b) * CB);
            for (size_t i = tid; i < (CB >> 2); i += 512) c4[i] = make_float4(0.f, 0.f, 0.f, 0.f);
            float4* n4 = (float4*)(pr.nst + ((size_t)l * BB + b) * NB);
            for (size_t i = tid; i < (NB >> 2); i += 512) n4[i] = make_float4(1.f, 1.f, 1.f, 1.f);
            if (tid < HH) pr.mst[((size_t)l * BB + b) * HH + tid] = 0.f;
        }
    }
    __syncthreads();

    for (int t = 0; t < SS; ++t) {
        // inp = x_t
        s_inp[tid] = pr.xbuf[((size_t)t * BB + b) * DD + tid];
        __syncthreads();

        for (int l = 0; l < LL; ++l) {
            // ---- LayerNorm over D ----
            float xi  = s_inp[tid];
            float mu  = blockSum512(xi, s_red, tid) * (1.0f / DD);
            float dif = xi - mu;
            float var = blockSum512(dif * dif, s_red, tid) * (1.0f / DD);
            s_xn[tid] = dif * rsqrtf(var + EPSF) * pr.ln_g[l * DD + tid] + pr.ln_b[l * DD + tid];
            __syncthreads();

            // ---- up projections ----
            gemv16(pr.w_ul + (size_t)l * PP * DD,  pr.b_ul + l * PP,  s_xn, s_xt, PP,  DD, wave, lane, 1.f);
            gemv16(pr.w_ur + (size_t)l * HDD * DD, pr.b_ur + l * HDD, s_xn, s_rt, HDD, DD, wave, lane, 1.f);
            __syncthreads();

            // ---- causal conv over feature axis + SiLU ----
            for (int j = tid; j < PP; j += 512) {
                float acc = pr.conv_b[l];
#pragma unroll
                for (int u = 0; u < KC; ++u) {
                    int jj = j - (KC - 1) + u;
                    if (jj >= 0) acc = fmaf(pr.conv_w[l * KC + u], s_xt[jj], acc);
                }
                s_xc[j] = acc * sigmoidf_(acc);
            }
            __syncthreads();

            // ---- gate / qkv projections ----
            gemv16(pr.w_q + (size_t)l * HDD * PP, pr.b_q + l * HDD, s_xc, s_q, HDD, PP, wave, lane, 1.f);
            gemv16(pr.w_k + (size_t)l * HDD * PP, pr.b_k + l * HDD, s_xc, s_k, HDD, PP, wave, lane, 0.125f);
            gemv16(pr.w_v + (size_t)l * HDD * PP, pr.b_v + l * HDD, s_xt, s_v, HDD, PP, wave, lane, 1.f);
            gemv16(pr.w_o + (size_t)l * HDD * PP, pr.b_o + l * HDD, s_xt, s_ot, HDD, PP, wave, lane, 1.f);
            {   // it (waves 0..7) / ft (waves 8..15), one wave per head
                const float4* Wp4 = (wave < 8)
                    ? (const float4*)(pr.w_i + ((size_t)l * HH + wave) * PP)
                    : (const float4*)(pr.w_f + ((size_t)l * HH + (wave - 8)) * PP);
                float bv = (wave < 8) ? pr.b_i[l * HH + wave] : pr.b_f[l * HH + (wave - 8)];
                const float4* xc4 = (const float4*)s_xc;
                float s = 0.f;
                for (int p = lane; p < (PP >> 2); p += 32) s = dot4(Wp4[p], xc4[p], s);
#pragma unroll
                for (int o = 16; o > 0; o >>= 1) s += __shfl_down(s, o, 32);
                if (lane == 0) {
                    if (wave < 8) s_it[wave] = s + bv; else s_ft[wave - 8] = s + bv;
                }
            }
            __syncthreads();

            // ---- output gate activation + head gate scalars ----
            s_ot[tid] = sigmoidf_(s_ot[tid]);
            if (tid < HH) {
                size_t mi = ((size_t)l * BB + b) * HH + tid;
                float mo = pr.mst[mi];
                float it = s_it[tid], ft = s_ft[tid];
                float mn = fmaxf(ft + mo, it);
                s_ie[tid] = __expf(it - mn);
                s_fe[tid] = __expf(ft - mn + mo);
                pr.mst[mi] = mn;
            }
            __syncthreads();

            // ---- state update: c, num, n, prods (float4 c traffic) ----
            {
                int h = tid >> 6, d = tid & 63;
                float fe = s_fe[h], ie = s_ie[h];
                float ievh = ie * s_v[h * DHH + d];
                float4* c4 = (float4*)(pr.cst + ((((size_t)l * BB + b) * HH + h) * DHH + d) * DHH);
                const float4* k4 = (const float4*)(s_k + h * DHH);
                const float4* q4 = (const float4*)(s_q + h * DHH);
                float accnum = 0.f;
#pragma unroll 4
                for (int p = 0; p < (DHH >> 2); ++p) {
                    float4 c = c4[p];
                    float4 kk = k4[p];
                    float4 qq = q4[p];
                    c.x = fmaf(ievh, kk.x, fe * c.x);
                    c.y = fmaf(ievh, kk.y, fe * c.y);
                    c.z = fmaf(ievh, kk.z, fe * c.z);
                    c.w = fmaf(ievh, kk.w, fe * c.w);
                    accnum = dot4(c, qq, accnum);
                    c4[p] = c;
                }
                s_num[tid] = accnum;
                // n update: interpret tid as (h, p)
                size_t ni = (((size_t)l * BB + b) * HH + h) * DHH + d;
                float nn = fmaf(ie, s_k[tid], fe * pr.nst[ni]);
                pr.nst[ni] = nn;
                s_xn[tid] = nn * s_q[tid];       // prods for den
            }
            __syncthreads();

            if (tid < HH) {
                float dsum = 0.f;
#pragma unroll 8
                for (int p = 0; p < DHH; ++p) dsum += s_xn[tid * DHH + p];
                s_den[tid] = fmaxf(fabsf(dsum), 1.0f);
            }
            __syncthreads();

            // ---- gated head output ----
            s_h[tid] = s_ot[tid] * s_num[tid] / s_den[tid >> 6];
            __syncthreads();

            // ---- GroupNorm stats per head ----
            if (tid < HH) {
                float m = 0.f;
#pragma unroll 8
                for (int p = 0; p < DHH; ++p) m += s_h[tid * DHH + p];
                m *= (1.0f / DHH);
                float v2 = 0.f;
#pragma unroll 8
                for (int p = 0; p < DHH; ++p) { float dd = s_h[tid * DHH + p] - m; v2 = fmaf(dd, dd, v2); }
                s_mu[tid]  = m;
                s_var[tid] = v2 * (1.0f / DHH);
            }
            __syncthreads();

            // ---- GN apply + skip projection + SiLU(rt) gate ----
            {
                const float4* xc4 = (const float4*)s_xc;
                for (int j = wave; j < HDD; j += 16) {
                    const float4* wr4 = (const float4*)(pr.skip_w + ((size_t)l * HDD + j) * PP);
                    float s = 0.f;
                    for (int p = lane; p < (PP >> 2); p += 32) s = dot4(wr4[p], xc4[p], s);
#pragma unroll
                    for (int o = 16; o > 0; o >>= 1) s += __shfl_down(s, o, 32);
                    if (lane == 0) {
                        int h = j >> 6;
                        float g = (s_h[j] - s_mu[h]) * rsqrtf(s_var[h] + EPSF)
                                  * pr.gn_g[l * HDD + j] + pr.gn_b[l * HDD + j];
                        float r = s_rt[j];
                        s_h[j] = (g + s) * (r * sigmoidf_(r));
                    }
                }
            }
            __syncthreads();

            // ---- down projection + residual ----
            {
                const float4* h4 = (const float4*)s_h;
                for (int j = wave; j < DD; j += 16) {
                    const float4* wr4 = (const float4*)(pr.w_dn + ((size_t)l * DD + j) * HDD);
                    float s = 0.f;
                    for (int p = lane; p < (HDD >> 2); p += 32) s = dot4(wr4[p], h4[p], s);
#pragma unroll
                    for (int o = 16; o > 0; o >>= 1) s += __shfl_down(s, o, 32);
                    if (lane == 0) s_inp[j] = s + pr.b_dn[l * DD + j] + s_inp[j];
                }
            }
            __syncthreads();
        } // layers

        pr.outbuf[((size_t)t * BB + b) * DD + tid] = s_inp[tid];
        __syncthreads();
    } // time

    // ---- write final (cT, nT, mT) into d_out tail ----
    {
        const size_t CB = (size_t)HH * DHH * DHH;
        const size_t NB = (size_t)HH * DHH;
        for (int l = 0; l < LL; ++l) {
            const float4* cs = (const float4*)(pr.cst + ((size_t)l * BB + b) * CB);
            float4*       cd = (float4*)(pr.out_c + ((size_t)l * BB + b) * CB);
            for (size_t i = tid; i < (CB >> 2); i += 512) cd[i] = cs[i];
            const float4* ns = (const float4*)(pr.nst + ((size_t)l * BB + b) * NB);
            float4*       nd = (float4*)(pr.out_n + ((size_t)l * BB + b) * NB);
            for (size_t i = tid; i < (NB >> 2); i += 512) nd[i] = ns[i];
            if (tid < HH) pr.out_m[((size_t)l * BB + b) * HH + tid] =
                              pr.mst[((size_t)l * BB + b) * HH + tid];
        }
    }
}

// ---------------- vocab projection: bf16 WMMA GEMM ----------------
// logits[M=1024][N=32000] = X[1024][512] * W[N][512]^T + bias
#define GT_BM 64
#define GT_BN 64
#define GT_BK 64
#define GT_PAD 4   // row stride 68 elems = 136 B -> 8 B aligned packed stores

__global__ __launch_bounds__(256) void logits_gemm(const float* __restrict__ X,
                                                   const float* __restrict__ Wv,
                                                   const float* __restrict__ bias,
                                                   float* __restrict__ Out) {
    __shared__ __bf16 As[GT_BM][GT_BK + GT_PAD];
    __shared__ __bf16 Bs[GT_BN][GT_BK + GT_PAD];

    const int tid  = threadIdx.x;
    const int n0   = blockIdx.x * GT_BN;
    const int m0   = blockIdx.y * GT_BM;
    const int wave = tid >> 5, lane = tid & 31;
    const int wy   = wave >> 2, wx = wave & 3;      // 2x4 wave grid
    const int hi   = lane >> 4, lp = lane & 15;

    v8f acc0 = {};
    v8f acc1 = {};

    for (int k0 = 0; k0 < DD; k0 += GT_BK) {
        // stage tiles: float4 global loads, 4 bf16 packed per 8-byte LDS store
#pragma unroll
        for (int it = 0; it < 4; ++it) {
            int idx = tid + it * 256;            // 0..1023
            int r   = idx >> 4;                  // 0..63
            int k4  = (idx & 15) << 2;           // 0..60 step 4
            float4 vA = *(const float4*)&X[(size_t)(m0 + r) * DD + k0 + k4];
            uint2 pa;
            pa.x = (unsigned)bf16_bits(vA.x) | ((unsigned)bf16_bits(vA.y) << 16);
            pa.y = (unsigned)bf16_bits(vA.z) | ((unsigned)bf16_bits(vA.w) << 16);
            *(uint2*)&As[r][k4] = pa;
            float4 vB = *(const float4*)&Wv[(size_t)(n0 + r) * DD + k0 + k4];
            uint2 pb;
            pb.x = (unsigned)bf16_bits(vB.x) | ((unsigned)bf16_bits(vB.y) << 16);
            pb.y = (unsigned)bf16_bits(vB.z) | ((unsigned)bf16_bits(vB.w) << 16);
            *(uint2*)&Bs[r][k4] = pb;
        }
        if (k0 + GT_BK < DD) {   // warm caches for next K tiles -> global_prefetch
            __builtin_prefetch(&X [(size_t)(m0 + (tid & 63)) * DD + k0 + GT_BK], 0, 1);
            __builtin_prefetch(&Wv[(size_t)(n0 + (tid & 63)) * DD + k0 + GT_BK], 0, 1);
        }
        __syncthreads();

#pragma unroll
        for (int ks = 0; ks < 2; ++ks) {
            const int kb = ks * 32;
            // B fragment (32x16, cols wx*16..): lane lp = column, hi-half takes K 16..31
            v16bf bfv;
#pragma unroll
            for (int p = 0; p < 8; ++p) {
                int kk = kb + hi * 16 + 2 * p;
                bfv[2 * p]     = Bs[wx * 16 + lp][kk];
                bfv[2 * p + 1] = Bs[wx * 16 + lp][kk + 1];
            }
            // two A fragments (rows wy*32 + s*16)
#pragma unroll
            for (int s = 0; s < 2; ++s) {
                int row = wy * 32 + s * 16 + lp;
                v16bf af;
#pragma unroll
                for (int p = 0; p < 8; ++p) {
                    int kk = kb + hi * 16 + 2 * p;
                    af[2 * p]     = As[row][kk];
                    af[2 * p + 1] = As[row][kk + 1];
                }
                if (s == 0)
                    acc0 = __builtin_amdgcn_wmma_f32_16x16x32_bf16(false, af, false, bfv,
                                                                   (short)0, acc0, false, false);
                else
                    acc1 = __builtin_amdgcn_wmma_f32_16x16x32_bf16(false, af, false, bfv,
                                                                   (short)0, acc1, false, false);
            }
        }
        __syncthreads();
    }

    // epilogue: C layout — VGPR r: row = r + hi*8; lane lp = col
    const int col = n0 + wx * 16 + lp;
    const float bv = bias[col];
#pragma unroll
    for (int r = 0; r < 8; ++r) {
        int m = m0 + wy * 32 + hi * 8 + r;
        Out[(size_t)m * VV + col] = acc0[r] + bv;
        m = m0 + wy * 32 + 16 + hi * 8 + r;
        Out[(size_t)m * VV + col] = acc1[r] + bv;
    }
}

// ---------------- in-place row softmax over V (float4 passes) ----------------
__global__ __launch_bounds__(256) void softmax_kernel(float* __restrict__ logits) {
    __shared__ float red[8];
    const int row  = blockIdx.x;
    float4* p4 = (float4*)(logits + (size_t)row * VV);
    const int NV4 = VV >> 2;   // 8000
    const int tid = threadIdx.x, lane = tid & 31, wave = tid >> 5;

    float mx = -3.0e38f;
    for (int i = tid; i < NV4; i += 256) {
        float4 v = p4[i];
        mx = fmaxf(mx, fmaxf(fmaxf(v.x, v.y), fmaxf(v.z, v.w)));
    }
#pragma unroll
    for (int o = 16; o > 0; o >>= 1) mx = fmaxf(mx, __shfl_down(mx, o, 32));
    if (lane == 0) red[wave] = mx;
    __syncthreads();
    if (tid == 0) {
        float m = red[0];
#pragma unroll
        for (int i = 1; i < 8; ++i) m = fmaxf(m, red[i]);
        red[0] = m;
    }
    __syncthreads();
    mx = red[0];
    __syncthreads();

    float sum = 0.f;
    for (int i = tid; i < NV4; i += 256) {
        float4 v = p4[i];
        v.x = __expf(v.x - mx); v.y = __expf(v.y - mx);
        v.z = __expf(v.z - mx); v.w = __expf(v.w - mx);
        p4[i] = v;
        sum += v.x + v.y + v.z + v.w;
    }
#pragma unroll
    for (int o = 16; o > 0; o >>= 1) sum += __shfl_down(sum, o, 32);
    if (lane == 0) red[wave] = sum;
    __syncthreads();
    if (tid == 0) {
        float s = 0.f;
#pragma unroll
        for (int i = 0; i < 8; ++i) s += red[i];
        red[0] = s;
    }
    __syncthreads();
    float inv = 1.0f / red[0];
    for (int i = tid; i < NV4; i += 256) {
        float4 v = p4[i];
        v.x *= inv; v.y *= inv; v.z *= inv; v.w *= inv;
        p4[i] = v;
    }
}

// ---------------- launcher ----------------
extern "C" void kernel_launch(void* const* d_in, const int* in_sizes, int n_in,
                              void* d_out, int out_size, void* d_ws, size_t ws_size,
                              hipStream_t stream) {
    (void)in_sizes; (void)n_in; (void)out_size; (void)ws_size;

    const int*   tokens = (const int*)  d_in[0];
    const float* emb    = (const float*)d_in[1];
    const float* ln_g   = (const float*)d_in[2];
    const float* ln_b   = (const float*)d_in[3];
    const float* gn_g   = (const float*)d_in[4];
    const float* gn_b   = (const float*)d_in[5];
    const float* w_ul   = (const float*)d_in[6];
    const float* b_ul   = (const float*)d_in[7];
    const float* w_ur   = (const float*)d_in[8];
    const float* b_ur   = (const float*)d_in[9];
    const float* w_dn   = (const float*)d_in[10];
    const float* b_dn   = (const float*)d_in[11];
    const float* conv_w = (const float*)d_in[12];
    const float* conv_b = (const float*)d_in[13];
    const float* skip_w = (const float*)d_in[14];
    const float* w_i    = (const float*)d_in[15];
    const float* b_i    = (const float*)d_in[16];
    const float* w_f    = (const float*)d_in[17];
    const float* b_f    = (const float*)d_in[18];
    const float* w_o    = (const float*)d_in[19];
    const float* b_o    = (const float*)d_in[20];
    const float* w_q    = (const float*)d_in[21];
    const float* b_q    = (const float*)d_in[22];
    const float* w_k    = (const float*)d_in[23];
    const float* b_k    = (const float*)d_in[24];
    const float* w_v    = (const float*)d_in[25];
    const float* b_v    = (const float*)d_in[26];
    const float* out_w  = (const float*)d_in[27];
    const float* out_b  = (const float*)d_in[28];

    float* ws     = (float*)d_ws;
    float* xbuf   = ws;                                         // S*B*D
    float* outbuf = xbuf   + (size_t)SS * BB * DD;              // S*B*D
    float* cst    = outbuf + (size_t)SS * BB * DD;              // L*B*H*Dh*Dh
    float* nst    = cst    + (size_t)LL * BB * HH * DHH * DHH;  // L*B*H*Dh
    float* mst    = nst    + (size_t)LL * BB * HH * DHH;        // L*B*H

    float* out    = (float*)d_out;
    float* probs  = out;
    float* out_c  = probs + (size_t)SS * BB * VV;
    float* out_n  = out_c + (size_t)LL * BB * HH * DHH * DHH;
    float* out_m  = out_n + (size_t)LL * BB * HH * DHH;

    embed_kernel<<<SS * BB, 128, 0, stream>>>(tokens, emb, xbuf);

    ScanParams sp;
    sp.ln_g = ln_g; sp.ln_b = ln_b; sp.gn_g = gn_g; sp.gn_b = gn_b;
    sp.w_ul = w_ul; sp.b_ul = b_ul; sp.w_ur = w_ur; sp.b_ur = b_ur;
    sp.w_dn = w_dn; sp.b_dn = b_dn;
    sp.conv_w = conv_w; sp.conv_b = conv_b; sp.skip_w = skip_w;
    sp.w_i = w_i; sp.b_i = b_i; sp.w_f = w_f; sp.b_f = b_f;
    sp.w_o = w_o; sp.b_o = b_o;
    sp.w_q = w_q; sp.b_q = b_q; sp.w_k = w_k; sp.b_k = b_k;
    sp.w_v = w_v; sp.b_v = b_v;
    sp.xbuf = xbuf; sp.outbuf = outbuf; sp.cst = cst; sp.nst = nst; sp.mst = mst;
    sp.out_c = out_c; sp.out_n = out_n; sp.out_m = out_m;

    scan_kernel<<<BB, 512, 0, stream>>>(sp);

    dim3 ggrid(VV / GT_BN, (SS * BB) / GT_BM);
    logits_gemm<<<ggrid, 256, 0, stream>>>(outbuf, out_w, out_b, probs);

    softmax_kernel<<<SS * BB, 256, 0, stream>>>(probs);
}
